// ESNtorch_37984690765942
// MI455X (gfx1250) — matrix operand order repaired and checked by reading, not compiled, not run on hardware
//
#include <hip/hip_runtime.h>
#include <hip/hip_bf16.h>

// ---------------------------------------------------------------------------
// ESN recurrence on gfx1250 (MI455X), wave32 + V_WMMA_F32_16X16X32_BF16.
// state_{t+1} = 0.7*state_t + 0.3*tanh(state_t @ W^T + x_t @ W_in^T)
// B=256, T=1024, D=128, R=1024.
// bf16x3 split precision (~fp32 accuracy); state kept in f32 AND as
// pre-split bf16 hi/lo buffers so the recurrent K-loop needs no conversions.
// ---------------------------------------------------------------------------

typedef __attribute__((ext_vector_type(16))) __bf16 v16bf;
typedef __attribute__((ext_vector_type(8)))  __bf16 v8bf;
typedef __attribute__((ext_vector_type(8)))  float  v8f;

#define BDIM 256
#define TDIM 1024
#define DDIM 128
#define RDIM 1024

// A fragment (16x32, M x K) straight from a row-major bf16 matrix.
// Layout (ISA 7.12.2, 16-bit A 16x32): lane L: M = L&15, h = L>>4;
//   a[0..7]  = row M, K = 8h .. 8h+7        (16B contiguous)
//   a[8..15] = row M, K = 16+8h .. 16+8h+7  (16B contiguous)
__device__ inline v16bf load_a_bf16(const __bf16* __restrict__ base, long ld,
                                    int row0, int k, int lane) {
  const int M = lane & 15;
  const int h = (lane >> 4) & 1;
  const __bf16* p = base + (long)(row0 + M) * ld + (k + 8 * h);
  v8bf a0 = *(const v8bf*)p;
  v8bf a1 = *(const v8bf*)(p + 16);
  v16bf r;
#pragma unroll
  for (int j = 0; j < 8; ++j) { r[j] = a0[j]; r[8 + j] = a1[j]; }
  return r;
}

struct ASplit { v16bf hi; v16bf lo; };

// A fragment from an f32 row-major matrix (used only for the x drive),
// split into hi/lo bf16 in-register.
__device__ inline ASplit load_a_split_f32(const float* __restrict__ base, long ld,
                                          int row0, int k, int lane) {
  const int M = lane & 15;
  const int h = (lane >> 4) & 1;
  const float* p = base + (long)(row0 + M) * ld + (k + 8 * h);
  const float4 f0 = *(const float4*)(p);
  const float4 f1 = *(const float4*)(p + 4);
  const float4 f2 = *(const float4*)(p + 16);
  const float4 f3 = *(const float4*)(p + 20);
  float v[16] = {f0.x, f0.y, f0.z, f0.w, f1.x, f1.y, f1.z, f1.w,
                 f2.x, f2.y, f2.z, f2.w, f3.x, f3.y, f3.z, f3.w};
  ASplit r;
#pragma unroll
  for (int j = 0; j < 16; ++j) {
    __bf16 hh = (__bf16)v[j];
    r.hi[j] = hh;
    r.lo[j] = (__bf16)(v[j] - (float)hh);
  }
  return r;
}

// B fragment (32x16, K x N) where B(k,n) = Wrow[n][k] (row-major bf16 source).
// lane L: N = L&15, h = L>>4; b[0..7] = column N, K = 8h..8h+7; b[8..15] = K = 16+8h..16+8h+7.
__device__ inline v16bf load_b16(const __bf16* __restrict__ base, long ld,
                                 int col0, int k, int lane) {
  const int N = lane & 15;
  const int h = (lane >> 4) & 1;
  const __bf16* p = base + (long)(col0 + N) * ld + (k + 8 * h);
  v8bf b0 = *(const v8bf*)p;
  v8bf b1 = *(const v8bf*)(p + 16);
  v16bf r;
#pragma unroll
  for (int j = 0; j < 8; ++j) { r[j] = b0[j]; r[8 + j] = b1[j]; }
  return r;
}

// 3-term split-precision MAC: (Ahi+Alo)(Bhi+Blo) ~= AhiBhi + AloBhi + AhiBlo
__device__ inline v8f wmma3(v16bf ahi, v16bf alo, v16bf bh, v16bf bl, v8f c) {
  c = __builtin_amdgcn_wmma_f32_16x16x32_bf16(false, ahi, false, bh, (short)0, c, false, false);
  c = __builtin_amdgcn_wmma_f32_16x16x32_bf16(false, alo, false, bh, (short)0, c, false, false);
  c = __builtin_amdgcn_wmma_f32_16x16x32_bf16(false, ahi, false, bl, (short)0, c, false, false);
  return c;
}

// One recurrence step. Grid: 32 blocks x 256 threads (8 wave32s) = 256 waves,
// each wave owns a 32x32 output block of the (256 x 1024) state.
__global__ __launch_bounds__(256) void esn_step(
    const float* __restrict__ Sprev,           // (256,1024) f32
    const __hip_bfloat16* __restrict__ Shi_,   // (256,1024) bf16 hi of Sprev
    const __hip_bfloat16* __restrict__ Slo_,   // (256,1024) bf16 lo of Sprev
    const float* __restrict__ xt,              // x + t*D; batch stride T*D
    const __hip_bfloat16* __restrict__ Whi_,   // (1024,1024) bf16 hi
    const __hip_bfloat16* __restrict__ Wlo_,   // (1024,1024) bf16 lo
    const __hip_bfloat16* __restrict__ WinHi_, // (1024,128) bf16 hi
    const __hip_bfloat16* __restrict__ WinLo_, // (1024,128) bf16 lo
    float* __restrict__ Snext,                 // (256,1024) f32
    __hip_bfloat16* __restrict__ SnHi_,        // bf16 hi of Snext
    __hip_bfloat16* __restrict__ SnLo_)        // bf16 lo of Snext
{
  const __bf16* Shi   = (const __bf16*)Shi_;
  const __bf16* Slo   = (const __bf16*)Slo_;
  const __bf16* Whi   = (const __bf16*)Whi_;
  const __bf16* Wlo   = (const __bf16*)Wlo_;
  const __bf16* WinHi = (const __bf16*)WinHi_;
  const __bf16* WinLo = (const __bf16*)WinLo_;
  __bf16* SnHi = (__bf16*)SnHi_;
  __bf16* SnLo = (__bf16*)SnLo_;

  const int lane = threadIdx.x & 31;
  const int wv   = (blockIdx.x << 3) | (threadIdx.x >> 5); // 0..255
  const int m0   = (wv & 7) << 5;   // 0..224
  const int n0   = (wv >> 3) << 5;  // 0..992

  v8f acc00 = {}, acc01 = {}, acc10 = {}, acc11 = {};

  // Recurrent part: K over R=1024 (state @ W^T), B(k,n)=W[n][k].
  // Pure bf16 b128 loads -> WMMA; no conversion VALU in this loop.
  for (int k = 0; k < RDIM; k += 32) {
    v16bf a0h = load_a_bf16(Shi, RDIM, m0,      k, lane);
    v16bf a0l = load_a_bf16(Slo, RDIM, m0,      k, lane);
    v16bf a1h = load_a_bf16(Shi, RDIM, m0 + 16, k, lane);
    v16bf a1l = load_a_bf16(Slo, RDIM, m0 + 16, k, lane);
    v16bf b0h = load_b16(Whi, RDIM, n0,      k, lane);
    v16bf b0l = load_b16(Wlo, RDIM, n0,      k, lane);
    v16bf b1h = load_b16(Whi, RDIM, n0 + 16, k, lane);
    v16bf b1l = load_b16(Wlo, RDIM, n0 + 16, k, lane);
    acc00 = wmma3(a0h, a0l, b0h, b0l, acc00);
    acc01 = wmma3(a0h, a0l, b1h, b1l, acc01);
    acc10 = wmma3(a1h, a1l, b0h, b0l, acc10);
    acc11 = wmma3(a1h, a1l, b1h, b1l, acc11);
  }

  // Input drive: K over D=128 (x_t @ W_in^T), fused into same accumulators.
  // x stays f32 in memory (too large to pre-split); split in-register.
  const long TD = (long)TDIM * DDIM;
  for (int k = 0; k < DDIM; k += 32) {
    ASplit a0 = load_a_split_f32(xt, TD, m0,      k, lane);
    ASplit a1 = load_a_split_f32(xt, TD, m0 + 16, k, lane);
    v16bf b0h = load_b16(WinHi, DDIM, n0,      k, lane);
    v16bf b0l = load_b16(WinLo, DDIM, n0,      k, lane);
    v16bf b1h = load_b16(WinHi, DDIM, n0 + 16, k, lane);
    v16bf b1l = load_b16(WinLo, DDIM, n0 + 16, k, lane);
    acc00 = wmma3(a0.hi, a0.lo, b0h, b0l, acc00);
    acc01 = wmma3(a0.hi, a0.lo, b1h, b1l, acc01);
    acc10 = wmma3(a1.hi, a1.lo, b0h, b0l, acc10);
    acc11 = wmma3(a1.hi, a1.lo, b1h, b1l, acc11);
  }

  // Epilogue: leaky tanh update + emit pre-split bf16 state for next step.
  // C/D layout: VGPR i, lane L: M = i + 8*(L>>4), N = L&15.
  const int nc = lane & 15;
  const int mh = (lane >> 4) ? 8 : 0;
#pragma unroll
  for (int ti = 0; ti < 2; ++ti) {
#pragma unroll
    for (int tj = 0; tj < 2; ++tj) {
      v8f a = (ti == 0) ? ((tj == 0) ? acc00 : acc01)
                        : ((tj == 0) ? acc10 : acc11);
#pragma unroll
      for (int i = 0; i < 8; ++i) {
        const int m = m0 + 16 * ti + i + mh;
        const int n = n0 + 16 * tj + nc;
        const long idx = (long)m * RDIM + n;
        const float s = 0.7f * Sprev[idx] + 0.3f * tanhf(a[i]);
        Snext[idx] = s;
        const __bf16 sh = (__bf16)s;
        SnHi[idx] = sh;
        SnLo[idx] = (__bf16)(s - (float)sh);
      }
    }
  }
}

// One-time per launch: split W and W_in into bf16 hi/lo pairs.
__global__ void esn_prep(const float* __restrict__ W, const float* __restrict__ Win,
                         __hip_bfloat16* Whi_, __hip_bfloat16* Wlo_,
                         __hip_bfloat16* WinHi_, __hip_bfloat16* WinLo_) {
  __bf16* Whi   = (__bf16*)Whi_;
  __bf16* Wlo   = (__bf16*)Wlo_;
  __bf16* WinHi = (__bf16*)WinHi_;
  __bf16* WinLo = (__bf16*)WinLo_;
  const long stride = (long)gridDim.x * blockDim.x;
  const long tid = (long)blockIdx.x * blockDim.x + threadIdx.x;
  for (long i = tid; i < (long)RDIM * RDIM; i += stride) {
    float f = W[i];
    __bf16 h = (__bf16)f;
    Whi[i] = h;
    Wlo[i] = (__bf16)(f - (float)h);
  }
  for (long i = tid; i < (long)RDIM * DDIM; i += stride) {
    float f = Win[i];
    __bf16 h = (__bf16)f;
    WinHi[i] = h;
    WinLo[i] = (__bf16)(f - (float)h);
  }
}

extern "C" void kernel_launch(void* const* d_in, const int* in_sizes, int n_in,
                              void* d_out, int out_size, void* d_ws, size_t ws_size,
                              hipStream_t stream) {
  (void)in_sizes; (void)n_in; (void)out_size; (void)ws_size;
  const float* x   = (const float*)d_in[0]; // (256,1024,128)
  const float* W   = (const float*)d_in[1]; // (1024,1024)
  const float* Win = (const float*)d_in[2]; // (1024,128)
  float* out = (float*)d_out;               // (256,1024)

  // Workspace layout (8.5 MB total), sizes in bytes:
  //   0x000000 S0    f32  1 MB     0x100000 Shi0 bf16 512 KB
  //   0x180000 Slo0  bf16 512 KB   0x200000 S1   f32  1 MB
  //   0x300000 Shi1  bf16 512 KB   0x380000 Slo1 bf16 512 KB
  //   0x400000 Whi   bf16 2 MB     0x600000 Wlo  bf16 2 MB
  //   0x800000 WinHi bf16 256 KB   0x840000 WinLo bf16 256 KB
  char* ws = (char*)d_ws;
  float* S0            = (float*)(ws + 0x000000);
  __hip_bfloat16* Shi0 = (__hip_bfloat16*)(ws + 0x100000);
  __hip_bfloat16* Slo0 = (__hip_bfloat16*)(ws + 0x180000);
  float* S1            = (float*)(ws + 0x200000);
  __hip_bfloat16* Shi1 = (__hip_bfloat16*)(ws + 0x300000);
  __hip_bfloat16* Slo1 = (__hip_bfloat16*)(ws + 0x380000);
  __hip_bfloat16* Whi   = (__hip_bfloat16*)(ws + 0x400000);
  __hip_bfloat16* Wlo   = (__hip_bfloat16*)(ws + 0x600000);
  __hip_bfloat16* WinHi = (__hip_bfloat16*)(ws + 0x800000);
  __hip_bfloat16* WinLo = (__hip_bfloat16*)(ws + 0x840000);

  // state0 = 0 (f32 + bf16 hi/lo are contiguous in the first 2 MB).
  hipMemsetAsync(ws, 0, 0x200000, stream);
  esn_prep<<<512, 256, 0, stream>>>(W, Win, Whi, Wlo, WinHi, WinLo);

  for (int t = 0; t < TDIM; ++t) {
    const int odd = t & 1;
    const float*           Sp  = odd ? S1   : S0;
    const __hip_bfloat16*  SpH = odd ? Shi1 : Shi0;
    const __hip_bfloat16*  SpL = odd ? Slo1 : Slo0;
    float*          Sn  = odd ? S0   : S1;
    __hip_bfloat16* SnH = odd ? Shi0 : Shi1;
    __hip_bfloat16* SnL = odd ? Slo0 : Slo1;
    if (t == TDIM - 1) Sn = out;  // bf16 outputs go to dead ping-pong buffers
    esn_step<<<32, 256, 0, stream>>>(Sp, SpH, SpL, x + (size_t)t * DDIM,
                                     Whi, Wlo, WinHi, WinLo, Sn, SnH, SnL);
  }
}
